// DownConv_15625091022859
// MI455X (gfx1250) — compile-verified, matrix operand order checked by reading, not run on hardware
//
#include <hip/hip_runtime.h>
#include <math.h>

typedef __attribute__((ext_vector_type(16))) _Float16 v16h;
typedef __attribute__((ext_vector_type(8)))  _Float16 h8;
typedef __attribute__((ext_vector_type(8)))  float    v8f;

#define BB       4
#define NPB      65536
#define NPTS     (BB*NPB)
#define CINC     32
#define COUTC    64
#define RIN      128
#define RR       64
#define R2       (RR*RR)
#define BN_EPS   1e-5f
#define PADC     0.1f
#define ORD_NEG_INF 0x007FFFFFu   // f2ord(-inf)

enum {
  IN_P = 0, IN_XXY, IN_XXZ, IN_XYZ, IN_C,
  P_C1A_W, P_C1A_B, P_C1A_G, P_C1A_BE,
  P_C1B_W, P_C1B_B, P_C1B_G, P_C1B_BE,
  P_K_W, P_K_B, P_V_W1, P_V_B1, P_V_W2, P_V_B2, P_V_W3, P_V_B3,
  P_Q_W, P_Q_B,
  P_POS_W1, P_POS_B1, P_POS_G, P_POS_BE, P_POS_W2, P_POS_B2,
  P_W_W, P_W_B, P_W_G, P_W_BE,
  P_WC_W, P_WC_B, P_CX_W, P_CX_B
};

__device__ __forceinline__ float normc(float v) {
  float x = v / (1.0f + PADC + 1e-3f) + 0.5f;
  return fminf(fmaxf(x, 0.0f), 1.0f - 1e-6f);
}
__device__ __forceinline__ unsigned f2ord(float f) {
  unsigned u = __float_as_uint(f);
  return (u & 0x80000000u) ? ~u : (u | 0x80000000u);
}
__device__ __forceinline__ float ord2f(unsigned u) {
  return (u & 0x80000000u) ? __uint_as_float(u & 0x7fffffffu) : __uint_as_float(~u);
}

// ---- WMMA GEMM: out[M,Nc] = act( A[M,K](f16) @ W[Nc,K]^T(f16) + bias )
// N-tiled: each wave computes a full 16 x Nc row-stripe (NT = Nc/16 accumulators),
// so the large A operand is loaded once per K-step instead of NT times.
// 4 waves per workgroup; wave w of block bx owns rows [ (bx*4+w)*16, +16 ).
template<int NT, bool RELU, bool OUT16>
__global__ __launch_bounds__(128)
void wmma_gemm_kernel(const _Float16* __restrict__ A, const _Float16* __restrict__ Wt,
                      const float* __restrict__ bias, float* __restrict__ outF,
                      _Float16* __restrict__ outH, int M, int K)
{
  const int lane = threadIdx.x & 31;
  const int wave = threadIdx.x >> 5;
  const int lr   = lane & 15;
  const int sel  = lane >> 4;         // 0: K {0..7,16..23}; 1: K {8..15,24..31}
  const int off  = sel * 8;
  const int rtile = blockIdx.x * 4 + wave;
  const int row   = rtile * 16 + lr;
  const int Nc    = NT * 16;

  const _Float16* __restrict__ arow = A + (size_t)row * K;

  v8f acc[NT];
#pragma unroll
  for (int nt = 0; nt < NT; ++nt) acc[nt] = v8f{};

  for (int kk = 0; kk < K; kk += 32) {
    h8 alo = *(const h8*)(arow + kk + off);
    h8 ahi = *(const h8*)(arow + kk + off + 16);
    v16h a;
#pragma unroll
    for (int j = 0; j < 8; ++j) { a[j] = alo[j]; a[8 + j] = ahi[j]; }
#pragma unroll
    for (int nt = 0; nt < NT; ++nt) {
      const _Float16* brow = Wt + (size_t)(nt * 16 + lr) * K + kk + off;
      h8 blo = *(const h8*)(brow);
      h8 bhi = *(const h8*)(brow + 16);
      v16h b;
#pragma unroll
      for (int j = 0; j < 8; ++j) { b[j] = blo[j]; b[8 + j] = bhi[j]; }
      acc[nt] = __builtin_amdgcn_wmma_f32_16x16x32_f16(false, a, false, b,
                                                       (short)0, acc[nt], false, false);
    }
  }

#pragma unroll
  for (int nt = 0; nt < NT; ++nt) {
    const int col = nt * 16 + lr;
    const float bv = bias ? bias[col] : 0.0f;
#pragma unroll
    for (int r = 0; r < 8; ++r) {
      const int orow = rtile * 16 + r + sel * 8;   // D layout: VGPR r -> row r / 8+r
      float v = acc[nt][r] + bv;
      if (RELU) v = fmaxf(v, 0.0f);
      if (OUT16) outH[(size_t)orow * Nc + col] = (_Float16)v;
      else       outF[(size_t)orow * Nc + col] = v;
    }
  }
}

__global__ void f32_to_f16_kernel(const float* __restrict__ s, _Float16* __restrict__ d, int n) {
  int i = blockIdx.x * blockDim.x + threadIdx.x;
  if (i < n) d[i] = (_Float16)s[i];
}
__global__ void fill_f32_kernel(float* __restrict__ d, float v, int n) {
  int i = blockIdx.x * blockDim.x + threadIdx.x;
  if (i < n) d[i] = v;
}
__global__ void fill_u32_kernel(unsigned* __restrict__ d, unsigned v, int n) {
  int i = blockIdx.x * blockDim.x + threadIdx.x;
  if (i < n) d[i] = v;
}
__global__ void copy_f32_kernel(const float* __restrict__ s, float* __restrict__ d, int n) {
  int i = blockIdx.x * blockDim.x + threadIdx.x;
  if (i < n) d[i] = s[i];
}

__global__ void maxpool2_kernel(const float* __restrict__ in, float* __restrict__ out) {
  int i = blockIdx.x * blockDim.x + threadIdx.x;
  const int n = BB * CINC * R2;
  if (i >= n) return;
  int x = i & (RR - 1), y = (i >> 6) & (RR - 1), bc = i >> 12;
  const float* p = in + ((size_t)bc * RIN + 2 * y) * RIN + 2 * x;
  out[i] = fmaxf(fmaxf(p[0], p[1]), fmaxf(p[RIN], p[RIN + 1]));
}

template<int CI, bool BN, bool RELU>
__global__ void conv3x3_kernel(const float* __restrict__ in, const float* __restrict__ w,
                               const float* __restrict__ bias, const float* __restrict__ g,
                               const float* __restrict__ be, float* __restrict__ out, int CO)
{
  int i = blockIdx.x * blockDim.x + threadIdx.x;
  const int total = BB * CO * R2;
  if (i >= total) return;
  int x = i & (RR - 1), y = (i >> 6) & (RR - 1);
  int co = (i >> 12) % CO;
  int b  = i / (CO * R2);
  float acc = bias[co];
  const float* wbase = w + (size_t)co * CI * 9;
  const float* ibase = in + (size_t)b * CI * R2;
  for (int ci = 0; ci < CI; ++ci) {
    const float* ip = ibase + (size_t)ci * R2;
    const float* wp = wbase + ci * 9;
#pragma unroll
    for (int ky = 0; ky < 3; ++ky) {
      int yy = y + ky - 1;
      if (yy < 0 || yy >= RR) continue;
#pragma unroll
      for (int kx = 0; kx < 3; ++kx) {
        int xx = x + kx - 1;
        if (xx < 0 || xx >= RR) continue;
        acc += wp[ky * 3 + kx] * ip[yy * RR + xx];
      }
    }
  }
  if (BN)   acc = acc * (g[co] * rsqrtf(1.0f + BN_EPS)) + be[co];
  if (RELU) acc = fmaxf(acc, 0.0f);
  out[i] = acc;
}

__global__ void outplane_kernel(const float* __restrict__ fea, const float* __restrict__ feat,
                                const float* __restrict__ wcw, const float* __restrict__ wcb,
                                const float* __restrict__ cxw, const float* __restrict__ cxb,
                                float* __restrict__ out)
{
  int i = blockIdx.x * blockDim.x + threadIdx.x;
  const int total = BB * COUTC * R2;
  if (i >= total) return;
  int x = i & (RR - 1), y = (i >> 6) & (RR - 1);
  int co = (i >> 12) & (COUTC - 1);
  int b  = i >> 18;
  float a0 = wcb[co], a1 = cxb[co];
  const float* w0 = wcw + (size_t)co * COUTC * 9;
  const float* w1 = cxw + (size_t)co * COUTC * 9;
  const float* i0 = fea  + (size_t)b * COUTC * R2;
  const float* i1 = feat + (size_t)b * COUTC * R2;
  for (int ci = 0; ci < COUTC; ++ci) {
    const float* p0 = i0 + (size_t)ci * R2;
    const float* p1 = i1 + (size_t)ci * R2;
    const float* q0 = w0 + ci * 9;
    const float* q1 = w1 + ci * 9;
#pragma unroll
    for (int ky = 0; ky < 3; ++ky) {
      int yy = y + ky - 1;
      if (yy < 0 || yy >= RR) continue;
#pragma unroll
      for (int kx = 0; kx < 3; ++kx) {
        int xx = x + kx - 1;
        if (xx < 0 || xx >= RR) continue;
        a0 += q0[ky * 3 + kx] * p0[yy * RR + xx];
        a1 += q1[ky * 3 + kx] * p1[yy * RR + xx];
      }
    }
  }
  out[i] = fmaxf(a0, 0.0f) + a1;
}

__global__ void feat_transpose16_kernel(const float* __restrict__ feat, _Float16* __restrict__ out) {
  int i = blockIdx.x * blockDim.x + threadIdx.x;
  const int n = BB * R2 * COUTC;
  if (i >= n) return;
  int c = i & (COUTC - 1);
  int pix = (i >> 6) & (R2 - 1);
  int b = i >> 18;
  out[i] = (_Float16)feat[((size_t)b * COUTC + c) * R2 + pix];
}

__global__ void pass1_kernel(const float* __restrict__ p, const float* __restrict__ kf,
                             const float* __restrict__ qt,
                             const float* __restrict__ pw1, const float* __restrict__ pb1,
                             const float* __restrict__ pg,  const float* __restrict__ pbe,
                             const float* __restrict__ pw2, const float* __restrict__ pb2,
                             const float* __restrict__ ww,  const float* __restrict__ wb,
                             const float* __restrict__ wg,  const float* __restrict__ wbe,
                             int c0, int c1,
                             int* __restrict__ idx_out, float* __restrict__ w8_out,
                             unsigned* __restrict__ wmax)
{
  int i = blockIdx.x * blockDim.x + threadIdx.x;
  if (i >= NPTS) return;
  int b = i >> 16;
  const float* pp = p + (size_t)i * 3;
  float u = normc(pp[c0]), v = normc(pp[c1]);
  float fx = floorf(u * RR), fy = floorf(v * RR);
  int bin = (int)fx + RR * (int)fy;
  float px = u - fx * (1.0f / RR), py = v - fy * (1.0f / RR);
  float h0 = pw1[0] * px + pw1[1] * py + pb1[0];
  float h1 = pw1[2] * px + pw1[3] * py + pb1[1];
  h0 = fmaxf(h0 * (pg[0] * rsqrtf(1.0f + BN_EPS)) + pbe[0], 0.0f);
  h1 = fmaxf(h1 * (pg[1] * rsqrtf(1.0f + BN_EPS)) + pbe[1], 0.0f);
  const float* kp = kf + (size_t)i * COUTC;
  const float* qp = qt + (size_t)(b * R2 + bin) * COUTC;
  float acc[8];
#pragma unroll
  for (int j = 0; j < 8; ++j) acc[j] = wb[j];
  for (int c = 0; c < COUTC; ++c) {
    float pe = pw2[2 * c] * h0 + pw2[2 * c + 1] * h1 + pb2[c];
    float wr = kp[c] - qp[c] + pe;
#pragma unroll
    for (int j = 0; j < 8; ++j) acc[j] += ww[j * COUTC + c] * wr;
  }
  idx_out[i] = bin;
  unsigned base = (unsigned)(b * R2 + bin) * 8u;
#pragma unroll
  for (int j = 0; j < 8; ++j) {
    float wv = fmaxf(acc[j] * (wg[j] * rsqrtf(1.0f + BN_EPS)) + wbe[j], 0.0f);
    w8_out[(size_t)i * 8 + j] = wv;
    atomicMax(&wmax[base + j], f2ord(wv));
  }
}

__global__ void pass2_kernel(const int* __restrict__ idx, float* __restrict__ w8,
                             const unsigned* __restrict__ wmax, float* __restrict__ wsum)
{
  int i = blockIdx.x * blockDim.x + threadIdx.x;
  if (i >= NPTS * 8) return;
  int j = i & 7, pt = i >> 3, b = pt >> 16;
  int s = (b * R2 + idx[pt]) * 8 + j;
  float e = __expf(w8[i] - ord2f(wmax[s]));
  w8[i] = e;
  atomicAdd(&wsum[s], e);
}

__global__ void pass3_kernel(const float* __restrict__ p, const float* __restrict__ vf,
                             const int* __restrict__ idx, const float* __restrict__ w8e,
                             const float* __restrict__ wsum,
                             const float* __restrict__ pw1, const float* __restrict__ pb1,
                             const float* __restrict__ pg,  const float* __restrict__ pbe,
                             const float* __restrict__ pw2, const float* __restrict__ pb2,
                             int c0, int c1, float* __restrict__ fea)
{
  int i = blockIdx.x * blockDim.x + threadIdx.x;
  if (i >= NPTS) return;
  int b = i >> 16;
  const float* pp = p + (size_t)i * 3;
  float u = normc(pp[c0]), v = normc(pp[c1]);
  float fx = floorf(u * RR), fy = floorf(v * RR);
  float px = u - fx * (1.0f / RR), py = v - fy * (1.0f / RR);
  float h0 = pw1[0] * px + pw1[1] * py + pb1[0];
  float h1 = pw1[2] * px + pw1[3] * py + pb1[1];
  h0 = fmaxf(h0 * (pg[0] * rsqrtf(1.0f + BN_EPS)) + pbe[0], 0.0f);
  h1 = fmaxf(h1 * (pg[1] * rsqrtf(1.0f + BN_EPS)) + pbe[1], 0.0f);
  int bin = idx[i];
  float wts[8];
#pragma unroll
  for (int j = 0; j < 8; ++j)
    wts[j] = w8e[(size_t)i * 8 + j] / (wsum[(b * R2 + bin) * 8 + j] + 1e-16f);
  const float* vp = vf + (size_t)i * COUTC;
  for (int c = 0; c < COUTC; ++c) {
    float pe = pw2[2 * c] * h0 + pw2[2 * c + 1] * h1 + pb2[c];
    atomicAdd(&fea[((size_t)(b * COUTC + c)) * R2 + bin], (vp[c] + pe) * wts[c & 7]);
  }
}

__global__ void sample_acc_kernel(const float* __restrict__ p, const float* __restrict__ plane,
                                  int c0, int c1, float* __restrict__ cout)
{
  int i = blockIdx.x * blockDim.x + threadIdx.x;
  if (i >= NPTS) return;
  int b = i >> 16;
  const float* pp = p + (size_t)i * 3;
  float u = normc(pp[c0]), v = normc(pp[c1]);
  float ix = u * (RR - 1), iy = v * (RR - 1);
  float fx = floorf(ix), fy = floorf(iy);
  float wx = ix - fx, wy = iy - fy;
  int x0 = min(max((int)fx, 0), RR - 1);
  int x1 = min(max((int)fx + 1, 0), RR - 1);
  int y0 = min(max((int)fy, 0), RR - 1);
  int y1 = min(max((int)fy + 1, 0), RR - 1);
  float w00 = (1 - wx) * (1 - wy), w01 = wx * (1 - wy), w10 = (1 - wx) * wy, w11 = wx * wy;
  const float* base = plane + (size_t)b * COUTC * R2;
  float* co = cout + (size_t)i * COUTC;
  for (int c = 0; c < COUTC; ++c) {
    const float* f = base + (size_t)c * R2;
    co[c] += f[y0 * RR + x0] * w00 + f[y0 * RR + x1] * w01
           + f[y1 * RR + x0] * w10 + f[y1 * RR + x1] * w11;
  }
}

extern "C" void kernel_launch(void* const* d_in, const int* in_sizes, int n_in,
                              void* d_out, int out_size, void* d_ws, size_t ws_size,
                              hipStream_t stream) {
  (void)in_sizes; (void)n_in; (void)out_size; (void)ws_size;
  const float* p  = (const float*)d_in[IN_P];
  const float* cf = (const float*)d_in[IN_C];
  const float* xsrc[3] = { (const float*)d_in[IN_XXY], (const float*)d_in[IN_XXZ], (const float*)d_in[IN_XYZ] };
  const int pc0[3] = {0, 0, 1}, pc1[3] = {1, 2, 2};

  char* ws = (char*)d_ws;  size_t off = 0;
  auto alloc = [&](size_t bytes) -> void* {
    void* r = ws + off; off += (bytes + 255) & ~(size_t)255; return r;
  };
  _Float16* kw16  = (_Float16*)alloc((size_t)COUTC * CINC * 2);
  _Float16* vw116 = (_Float16*)alloc((size_t)COUTC * CINC * 2);
  _Float16* vw216 = (_Float16*)alloc((size_t)2 * COUTC * COUTC * 2);
  _Float16* vw316 = (_Float16*)alloc((size_t)COUTC * 2 * COUTC * 2);
  _Float16* qw16  = (_Float16*)alloc((size_t)COUTC * COUTC * 2);
  _Float16* A16   = (_Float16*)alloc((size_t)NPTS * CINC * 2);
  _Float16* H116  = (_Float16*)alloc((size_t)NPTS * COUTC * 2);
  _Float16* H216  = (_Float16*)alloc((size_t)NPTS * 2 * COUTC * 2);
  float* Kf     = (float*)alloc((size_t)NPTS * COUTC * 4);
  float* Vf     = (float*)alloc((size_t)NPTS * COUTC * 4);
  float* pooled = (float*)alloc((size_t)BB * CINC * R2 * 4);
  float* t1     = (float*)alloc((size_t)BB * COUTC * R2 * 4);
  float* feat   = (float*)alloc((size_t)BB * COUTC * R2 * 4);
  _Float16* featT16 = (_Float16*)alloc((size_t)BB * R2 * COUTC * 2);
  float* qT     = (float*)alloc((size_t)BB * R2 * COUTC * 4);
  int*   idxB   = (int*)alloc((size_t)NPTS * 4);
  float* w8B    = (float*)alloc((size_t)NPTS * 8 * 4);
  unsigned* wmaxB = (unsigned*)alloc((size_t)BB * R2 * 8 * 4);
  float* wsumB  = (float*)alloc((size_t)BB * R2 * 8 * 4);
  float* feaB   = (float*)alloc((size_t)BB * COUTC * R2 * 4);

  float* outPlane0 = (float*)d_out;
  float* coutP     = (float*)d_out + (size_t)3 * BB * COUTC * R2;

  const int T = 256;
  auto blk = [](int n, int t) { return (n + t - 1) / t; };

  f32_to_f16_kernel<<<blk(COUTC*CINC, T), T, 0, stream>>>((const float*)d_in[P_K_W],  kw16,  COUTC*CINC);
  f32_to_f16_kernel<<<blk(COUTC*CINC, T), T, 0, stream>>>((const float*)d_in[P_V_W1], vw116, COUTC*CINC);
  f32_to_f16_kernel<<<blk(2*COUTC*COUTC, T), T, 0, stream>>>((const float*)d_in[P_V_W2], vw216, 2*COUTC*COUTC);
  f32_to_f16_kernel<<<blk(2*COUTC*COUTC, T), T, 0, stream>>>((const float*)d_in[P_V_W3], vw316, 2*COUTC*COUTC);
  f32_to_f16_kernel<<<blk(COUTC*COUTC, T), T, 0, stream>>>((const float*)d_in[P_Q_W],  qw16,  COUTC*COUTC);
  f32_to_f16_kernel<<<blk(NPTS*CINC, T), T, 0, stream>>>(cf, A16, NPTS*CINC);

  // Each block = 4 waves; each wave computes a full 16 x Nc row-stripe.
  // k = cf @ k_w^T + k_b                         (M=NPTS, K=32,  Nc=64)
  wmma_gemm_kernel<4, false, false><<<NPTS/64, 128, 0, stream>>>(
      A16, kw16, (const float*)d_in[P_K_B], Kf, nullptr, NPTS, CINC);
  // h1 = relu(cf @ v_w1^T + b)                   (M=NPTS, K=32,  Nc=64)
  wmma_gemm_kernel<4, true, true><<<NPTS/64, 128, 0, stream>>>(
      A16, vw116, (const float*)d_in[P_V_B1], nullptr, H116, NPTS, CINC);
  // h2 = relu(h1 @ v_w2^T + b)                   (M=NPTS, K=64,  Nc=128)
  wmma_gemm_kernel<8, true, true><<<NPTS/64, 128, 0, stream>>>(
      H116, vw216, (const float*)d_in[P_V_B2], nullptr, H216, NPTS, COUTC);
  // v = h2 @ v_w3^T + b                          (M=NPTS, K=128, Nc=64)
  wmma_gemm_kernel<4, false, false><<<NPTS/64, 128, 0, stream>>>(
      H216, vw316, (const float*)d_in[P_V_B3], Vf, nullptr, NPTS, 2*COUTC);

  copy_f32_kernel<<<blk(NPTS*COUTC, T), T, 0, stream>>>(Vf, coutP, NPTS*COUTC);

  for (int pl = 0; pl < 3; ++pl) {
    float* outPlane = outPlane0 + (size_t)pl * BB * COUTC * R2;

    maxpool2_kernel<<<blk(BB*CINC*R2, T), T, 0, stream>>>(xsrc[pl], pooled);
    conv3x3_kernel<CINC, true, true><<<blk(BB*COUTC*R2, T), T, 0, stream>>>(
        pooled, (const float*)d_in[P_C1A_W], (const float*)d_in[P_C1A_B],
        (const float*)d_in[P_C1A_G], (const float*)d_in[P_C1A_BE], t1, COUTC);
    conv3x3_kernel<COUTC, true, true><<<blk(BB*COUTC*R2, T), T, 0, stream>>>(
        t1, (const float*)d_in[P_C1B_W], (const float*)d_in[P_C1B_B],
        (const float*)d_in[P_C1B_G], (const float*)d_in[P_C1B_BE], feat, COUTC);

    // q = 1x1 conv over pixels via WMMA GEMM     (M=B*R2=16384, K=64, Nc=64)
    feat_transpose16_kernel<<<blk(BB*R2*COUTC, T), T, 0, stream>>>(feat, featT16);
    wmma_gemm_kernel<4, false, false><<<(BB*R2)/64, 128, 0, stream>>>(
        featT16, qw16, (const float*)d_in[P_Q_B], qT, nullptr, BB*R2, COUTC);

    fill_u32_kernel<<<blk(BB*R2*8, T), T, 0, stream>>>(wmaxB, ORD_NEG_INF, BB*R2*8);
    fill_f32_kernel<<<blk(BB*R2*8, T), T, 0, stream>>>(wsumB, 0.0f, BB*R2*8);
    fill_f32_kernel<<<blk(BB*COUTC*R2, T), T, 0, stream>>>(feaB, 0.0f, BB*COUTC*R2);

    pass1_kernel<<<blk(NPTS, T), T, 0, stream>>>(
        p, Kf, qT,
        (const float*)d_in[P_POS_W1], (const float*)d_in[P_POS_B1],
        (const float*)d_in[P_POS_G],  (const float*)d_in[P_POS_BE],
        (const float*)d_in[P_POS_W2], (const float*)d_in[P_POS_B2],
        (const float*)d_in[P_W_W], (const float*)d_in[P_W_B],
        (const float*)d_in[P_W_G], (const float*)d_in[P_W_BE],
        pc0[pl], pc1[pl], idxB, w8B, wmaxB);
    pass2_kernel<<<blk(NPTS*8, T), T, 0, stream>>>(idxB, w8B, wmaxB, wsumB);
    pass3_kernel<<<blk(NPTS, T), T, 0, stream>>>(
        p, Vf, idxB, w8B, wsumB,
        (const float*)d_in[P_POS_W1], (const float*)d_in[P_POS_B1],
        (const float*)d_in[P_POS_G],  (const float*)d_in[P_POS_BE],
        (const float*)d_in[P_POS_W2], (const float*)d_in[P_POS_B2],
        pc0[pl], pc1[pl], feaB);

    outplane_kernel<<<blk(BB*COUTC*R2, T), T, 0, stream>>>(
        feaB, feat, (const float*)d_in[P_WC_W], (const float*)d_in[P_WC_B],
        (const float*)d_in[P_CX_W], (const float*)d_in[P_CX_B], outPlane);

    sample_acc_kernel<<<blk(NPTS, T), T, 0, stream>>>(p, outPlane, pc0[pl], pc1[pl], coutP);
  }
}
